// KTM_37761352466925
// MI455X (gfx1250) — matrix operand order repaired and checked by reading, not compiled, not run on hardware
//
#include <hip/hip_runtime.h>
#include <hip/hip_bf16.h>
#include <math.h>

// ---- problem constants (from reference) ----
#define B_   8
#define C1_  256
#define C2_  512
#define N_   4096   // H*W
#define M_   1024   // (H/2)*(W/2)
#define D_   64

typedef __bf16 bf16;
typedef __attribute__((ext_vector_type(16))) __bf16 bf16x16;
typedef __attribute__((ext_vector_type(8)))  __bf16 bf16x8;
typedef __attribute__((ext_vector_type(8)))  float  f32x8;

static __device__ __forceinline__ f32x8 zero8() {
  f32x8 z = {0.f, 0.f, 0.f, 0.f, 0.f, 0.f, 0.f, 0.f};
  return z;
}

// Load a 16-element bf16 fragment as two contiguous 16-byte chunks (b128 loads).
static __device__ __forceinline__ bf16x16 frag16(const bf16* p0, const bf16* p1) {
  bf16x8 a = *(const bf16x8*)p0;
  bf16x8 b = *(const bf16x8*)p1;
  bf16x16 r;
#pragma unroll
  for (int i = 0; i < 8; ++i) { r[i] = a[i]; r[i + 8] = b[i]; }
  return r;
}

static __device__ __forceinline__ f32x8 wmma_bf16(bf16x16 a, bf16x16 b, f32x8 c) {
  // D = A(16x32) * B(32x16) + C, f32 accumulate
  return __builtin_amdgcn_wmma_f32_16x16x32_bf16(false, a, false, b, (short)0, c,
                                                 false, false);
}

// ---------------------------------------------------------------------------
// Kernel 1: convert weights f32 -> bf16 (tiny, one shot per launch)
// ---------------------------------------------------------------------------
__global__ __launch_bounds__(256) void convert_weights_kernel(
    const float* __restrict__ wq, const float* __restrict__ wk,
    const float* __restrict__ wv, const float* __restrict__ wo,
    bf16* __restrict__ dq, bf16* __restrict__ dk,
    bf16* __restrict__ dv, bf16* __restrict__ dwo) {
  int i = blockIdx.x * 256 + threadIdx.x;       // grid covers 32768
  if (i < D_ * C1_) dq[i]  = (bf16)wq[i];
  if (i < D_ * C2_) { dk[i] = (bf16)wk[i]; dv[i] = (bf16)wv[i]; }
  if (i < C1_ * D_) dwo[i] = (bf16)wo[i];
}

// ---------------------------------------------------------------------------
// Kernel 2: 2x2 mean pool of x2 [B,C2,64,64] -> bf16 x2p [b][m][c2]
// (m-major layout so K/V projection B-fragments are row-contiguous in c)
// ---------------------------------------------------------------------------
__global__ __launch_bounds__(256) void pool_kernel(const float* __restrict__ x2,
                                                   bf16* __restrict__ x2p) {
  int idx = blockIdx.x * 256 + threadIdx.x;     // B_*C2_*M_ threads
  int m = idx & (M_ - 1);
  int c = (idx >> 10) & (C2_ - 1);
  int b = idx >> 19;
  int h2 = m >> 5, w2 = m & 31;
  const float* p = x2 + (((size_t)(b * C2_ + c) * 64 + h2 * 2) * 64 + w2 * 2);
  float s = (p[0] + p[1] + p[64] + p[65]) * 0.25f;
  x2p[((size_t)(b * M_ + m)) * C2_ + c] = (bf16)s;
}

// ---------------------------------------------------------------------------
// Kernel 3: Q projection. Q[n,d] = sum_c x1[b,c,n] * Wq[d,c] + bq[d]
// WMMA: A = Wq rows (d x c), B = x1 tile (c x n, lanes = n; the strided
// per-lane f32 loads are coalesced across lanes and the compiler turns them
// into b32 clauses + v_cvt_pk_bf16_f32). C tile stored transposed -> Qw[n][d].
// ---------------------------------------------------------------------------
__global__ __launch_bounds__(256) void proj_q_kernel(const float* __restrict__ x1,
                                                     const bf16* __restrict__ Wqb,
                                                     const float* __restrict__ bq,
                                                     bf16* __restrict__ Qw) {
  int tid = threadIdx.x, lane = tid & 31, wave = tid >> 5;
  int half = lane >> 4, l16 = lane & 15;
  int b  = blockIdx.x >> 7;                     // 128 blocks per batch
  int nb = (blockIdx.x & 127) * 32;
  int nt = wave & 1, dt = wave >> 1;            // 2 n-tiles x 4 d-tiles = 8 waves
  int n  = nb + nt * 16 + l16;

  f32x8 acc = zero8();
  for (int c0 = 0; c0 < C1_; c0 += 32) {
    const bf16* wp = Wqb + (dt * 16 + l16) * C1_ + c0 + half * 16;
    bf16x16 af = frag16(wp, wp + 8);
    bf16x16 bx;
    const float* xp = x1 + ((size_t)(b * C1_) + c0 + half * 16) * N_ + n;
#pragma unroll
    for (int e = 0; e < 16; ++e) bx[e] = (bf16)xp[(size_t)e * N_];
    acc = wmma_bf16(af, bx, acc);
  }
#pragma unroll
  for (int v = 0; v < 8; ++v) {
    int d = dt * 16 + v + half * 8;
    Qw[((size_t)(b * N_ + n)) * D_ + d] = (bf16)(acc[v] + bq[d]);
  }
}

// ---------------------------------------------------------------------------
// Kernel 4: K and V projections (share the x2p B fragment).
// Stored transposed to [b][m][d] bf16 (attention-friendly layouts).
// ---------------------------------------------------------------------------
__global__ __launch_bounds__(256) void proj_kv_kernel(const bf16* __restrict__ x2p,
                                                      const bf16* __restrict__ Wkb,
                                                      const bf16* __restrict__ Wvb,
                                                      const float* __restrict__ bk,
                                                      const float* __restrict__ bv,
                                                      bf16* __restrict__ Kw,
                                                      bf16* __restrict__ Vw) {
  int tid = threadIdx.x, lane = tid & 31, wave = tid >> 5;
  int half = lane >> 4, l16 = lane & 15;
  int b  = blockIdx.x >> 5;                     // 32 blocks per batch
  int mb = (blockIdx.x & 31) * 32;
  int mt = wave & 1, dt = wave >> 1;
  int m  = mb + mt * 16 + l16;

  f32x8 ka = zero8(), va = zero8();
  for (int c0 = 0; c0 < C2_; c0 += 32) {
    const bf16* wkp = Wkb + (dt * 16 + l16) * C2_ + c0 + half * 16;
    const bf16* wvp = Wvb + (dt * 16 + l16) * C2_ + c0 + half * 16;
    bf16x16 awk = frag16(wkp, wkp + 8);
    bf16x16 awv = frag16(wvp, wvp + 8);
    const bf16* xp = x2p + ((size_t)(b * M_ + m)) * C2_ + c0 + half * 16;
    bf16x16 bx = frag16(xp, xp + 8);
    ka = wmma_bf16(awk, bx, ka);
    va = wmma_bf16(awv, bx, va);
  }
#pragma unroll
  for (int v = 0; v < 8; ++v) {
    int d = dt * 16 + v + half * 8;
    size_t oi = ((size_t)(b * M_ + m)) * D_ + d;
    Kw[oi] = (bf16)(ka[v] + bk[d]);
    Vw[oi] = (bf16)(va[v] + bv[d]);
  }
}

// ---------------------------------------------------------------------------
// Kernel 5: flash attention. Block = 8 waves x 16 Q rows = 128 rows.
// Pipelined: chunk i+1 is fetched global->registers while chunk i's WMMAs run;
// the LOADcnt wait lands at the next iteration's LDS publish, so L2/HBM
// latency is hidden behind 8 WMMAs + softmax per chunk.
// K chunk staged [m][d] (stride 72), V chunk staged transposed [d][m]
// (stride 40); P goes through wave-private LDS for C->A layout conversion
// (in-order DS ops within a wave make this safe without a barrier).
// ---------------------------------------------------------------------------
__global__ __launch_bounds__(256) void attn_kernel(const bf16* __restrict__ Qw,
                                                   const bf16* __restrict__ Kw,
                                                   const bf16* __restrict__ Vw,
                                                   bf16* __restrict__ Ow) {
  __shared__ bf16 Klds[32 * 72];
  __shared__ bf16 Vtlds[64 * 40];
  __shared__ bf16 Plds[8 * 16 * 32];

  const int tid = threadIdx.x, lane = tid & 31, wave = tid >> 5;
  const int half = lane >> 4, l16 = lane & 15;
  const int b  = blockIdx.x >> 5;               // 32 blocks per batch
  const int nb = (blockIdx.x & 31) * 128;
  const int nrow = nb + wave * 16 + l16;

  // Q tile A-fragments held in registers for the whole M loop
  const bf16* qp = Qw + ((size_t)(b * N_ + nrow)) * D_ + half * 16;
  bf16x16 qf0 = frag16(qp, qp + 8);             // K(d) = 0..31
  bf16x16 qf1 = frag16(qp + 32, qp + 40);       // K(d) = 32..63

  // staging assignments (fixed per thread)
  const int krow = tid >> 3, kcol = (tid & 7) * 8;   // K: 32 rows x 64 cols
  const int vmm = tid & 31, vd0 = (tid >> 5) * 8;    // V: transpose 32x64
  const bf16* kbase = Kw + ((size_t)b * M_) * D_;
  const bf16* vbase = Vw + ((size_t)b * M_) * D_;

  // prefetch chunk 0 into registers
  bf16x8 kreg = *(const bf16x8*)(kbase + (size_t)krow * D_ + kcol);
  bf16x8 vreg = *(const bf16x8*)(vbase + (size_t)vmm * D_ + vd0);

  f32x8 O[4] = {zero8(), zero8(), zero8(), zero8()};
  float mrun[8], lrun[8];
#pragma unroll
  for (int v = 0; v < 8; ++v) { mrun[v] = -1e30f; lrun[v] = 0.f; }

  for (int m0 = 0; m0 < M_; m0 += 32) {
    __syncthreads();                 // all waves done reading previous chunk
    // publish prefetched chunk to LDS
    *(bf16x8*)&Klds[krow * 72 + kcol] = kreg;
#pragma unroll
    for (int i = 0; i < 8; ++i) Vtlds[(vd0 + i) * 40 + vmm] = vreg[i];
    __syncthreads();                 // chunk visible to all waves

    // issue next chunk's global loads; they complete during the WMMAs below
    if (m0 + 32 < M_) {
      kreg = *(const bf16x8*)(kbase + (size_t)(m0 + 32 + krow) * D_ + kcol);
      vreg = *(const bf16x8*)(vbase + (size_t)(m0 + 32 + vmm) * D_ + vd0);
    }

    // S = Q * K^T for two 16-column subtiles (K-dim = d, 2 WMMAs each)
    f32x8 S[2];
#pragma unroll
    for (int mt = 0; mt < 2; ++mt) {
      const bf16* kp = &Klds[(mt * 16 + l16) * 72 + half * 16];
      bf16x16 kf0 = frag16(kp, kp + 8);
      bf16x16 kf1 = frag16(kp + 32, kp + 40);
      f32x8 s = zero8();
      s = wmma_bf16(qf0, kf0, s);
      s = wmma_bf16(qf1, kf1, s);
      S[mt] = s;
    }

    // online softmax: each row lives across the 16 lanes of one half
#pragma unroll
    for (int v = 0; v < 8; ++v) {
      float x = fmaxf(S[0][v], S[1][v]);
      x = fmaxf(x, __shfl_xor(x, 1));
      x = fmaxf(x, __shfl_xor(x, 2));
      x = fmaxf(x, __shfl_xor(x, 4));
      x = fmaxf(x, __shfl_xor(x, 8));
      float nm    = fmaxf(mrun[v], x);
      float alpha = __expf(mrun[v] - nm);
      mrun[v] = nm;
      float p0 = __expf(S[0][v] - nm);
      float p1 = __expf(S[1][v] - nm);
      float rs = p0 + p1;
      rs += __shfl_xor(rs, 1);
      rs += __shfl_xor(rs, 2);
      rs += __shfl_xor(rs, 4);
      rs += __shfl_xor(rs, 8);
      lrun[v] = lrun[v] * alpha + rs;
      int row = v + half * 8;
      Plds[wave * 512 + row * 32 + l16]      = (bf16)p0;
      Plds[wave * 512 + row * 32 + 16 + l16] = (bf16)p1;
#pragma unroll
      for (int dt = 0; dt < 4; ++dt) O[dt][v] *= alpha;
    }

    // P (C layout) -> A fragment via wave-private LDS (DS ops in-order per wave)
    const bf16* pp = &Plds[wave * 512 + l16 * 32 + half * 16];
    bf16x16 pf = frag16(pp, pp + 8);
#pragma unroll
    for (int dt = 0; dt < 4; ++dt) {
      const bf16* vp = &Vtlds[(dt * 16 + l16) * 40 + half * 16];
      bf16x16 vf = frag16(vp, vp + 8);
      O[dt] = wmma_bf16(pf, vf, O[dt]);
    }
  }

  // epilogue: normalize and store O -> [b][n][d] bf16
#pragma unroll
  for (int v = 0; v < 8; ++v) {
    float inv = 1.f / lrun[v];
    int n = nb + wave * 16 + v + half * 8;
    bf16* dst = Ow + ((size_t)(b * N_ + n)) * D_ + l16;
#pragma unroll
    for (int dt = 0; dt < 4; ++dt) dst[dt * 16] = (bf16)(O[dt][v] * inv);
  }
}

// ---------------------------------------------------------------------------
// Kernel 6: output projection + bias + residual.
// y[b,c,n] = sum_d O[n,d]*Wo[c,d] + bo[c] + x1[b,c,n]
// ---------------------------------------------------------------------------
__global__ __launch_bounds__(256) void out_proj_kernel(const bf16* __restrict__ Ow,
                                                       const bf16* __restrict__ Wob,
                                                       const float* __restrict__ bo,
                                                       const float* __restrict__ x1,
                                                       float* __restrict__ y) {
  int tid = threadIdx.x, lane = tid & 31, wave = tid >> 5;
  int half = lane >> 4, l16 = lane & 15;
  int b  = blockIdx.x >> 7;                     // 128 blocks per batch
  int nb = (blockIdx.x & 127) * 32;
  int nt = wave & 1, cb = (wave >> 1) * 4;      // each wave: 4 c-tiles
  int n  = nb + nt * 16 + l16;

  f32x8 acc[4] = {zero8(), zero8(), zero8(), zero8()};
  for (int k0 = 0; k0 < D_; k0 += 32) {
    const bf16* op = Ow + ((size_t)(b * N_ + n)) * D_ + k0 + half * 16;
    bf16x16 bfr = frag16(op, op + 8);
#pragma unroll
    for (int j = 0; j < 4; ++j) {
      const bf16* wp = Wob + ((cb + j) * 16 + l16) * D_ + k0 + half * 16;
      bf16x16 af = frag16(wp, wp + 8);
      acc[j] = wmma_bf16(af, bfr, acc[j]);
    }
  }
#pragma unroll
  for (int j = 0; j < 4; ++j) {
#pragma unroll
    for (int v = 0; v < 8; ++v) {
      int c = (cb + j) * 16 + v + half * 8;
      size_t oi = ((size_t)(b * C1_ + c)) * N_ + n;
      y[oi] = acc[j][v] + bo[c] + x1[oi];
    }
  }
}

// ---------------------------------------------------------------------------
// Host launcher
// ---------------------------------------------------------------------------
extern "C" void kernel_launch(void* const* d_in, const int* in_sizes, int n_in,
                              void* d_out, int out_size, void* d_ws, size_t ws_size,
                              hipStream_t stream) {
  (void)in_sizes; (void)n_in; (void)out_size; (void)ws_size;
  const float* x1 = (const float*)d_in[0];
  const float* x2 = (const float*)d_in[1];
  const float* Wq = (const float*)d_in[2];
  const float* bq = (const float*)d_in[3];
  const float* Wk = (const float*)d_in[4];
  const float* bk = (const float*)d_in[5];
  const float* Wv = (const float*)d_in[6];
  const float* bv = (const float*)d_in[7];
  const float* Wo = (const float*)d_in[8];
  const float* bo = (const float*)d_in[9];
  float* out = (float*)d_out;

  // workspace layout (bf16 intermediates, ~18.3 MB, all L2-resident)
  char* ws = (char*)d_ws;
  bf16* Qw  = (bf16*)(ws);                                  // 4 MiB  [b][n][d]
  bf16* Kwp = (bf16*)(ws + ((size_t)4  << 20));             // 1 MiB  [b][m][d]
  bf16* Vwp = (bf16*)(ws + ((size_t)5  << 20));             // 1 MiB  [b][m][d]
  bf16* Ow  = (bf16*)(ws + ((size_t)6  << 20));             // 4 MiB  [b][n][d]
  bf16* X2p = (bf16*)(ws + ((size_t)10 << 20));             // 8 MiB  [b][m][c]
  bf16* Wqb = (bf16*)(ws + ((size_t)18 << 20));             // 32 KiB
  bf16* Wkb = (bf16*)(ws + ((size_t)18 << 20) + 65536);     // 64 KiB
  bf16* Wvb = (bf16*)(ws + ((size_t)18 << 20) + 131072);    // 64 KiB
  bf16* Wob = (bf16*)(ws + ((size_t)18 << 20) + 196608);    // 32 KiB

  convert_weights_kernel<<<128, 256, 0, stream>>>(Wq, Wk, Wv, Wo,
                                                  Wqb, Wkb, Wvb, Wob);
  pool_kernel<<<(B_ * C2_ * M_) / 256, 256, 0, stream>>>(x2, X2p);
  proj_q_kernel<<<B_ * (N_ / 32), 256, 0, stream>>>(x1, Wqb, bq, Qw);
  proj_kv_kernel<<<B_ * (M_ / 32), 256, 0, stream>>>(X2p, Wkb, Wvb, bk, bv,
                                                     Kwp, Vwp);
  attn_kernel<<<B_ * (N_ / 128), 256, 0, stream>>>(Qw, Kwp, Vwp, Ow);
  out_proj_kernel<<<B_ * (N_ / 32), 256, 0, stream>>>(Ow, Wob, bo, x1, out);
}
